// MultiHeadAttention_33122787786752
// MI455X (gfx1250) — matrix operand order repaired and checked by reading, not compiled, not run on hardware
//
#include <hip/hip_runtime.h>
#include <hip/hip_bf16.h>
#include <stdint.h>

// ---------------------------------------------------------------------------
// MHA forward for MI455X (gfx1250): f16 WMMA everywhere, flash-attention
// fusion (scores never touch HBM), async-to-LDS double-buffered K/V staging.
// ---------------------------------------------------------------------------

typedef __attribute__((ext_vector_type(16))) _Float16 v16h;
typedef __attribute__((ext_vector_type(8)))  _Float16 v8h;
typedef __attribute__((ext_vector_type(4)))  _Float16 v4h;
typedef __attribute__((ext_vector_type(8)))  float    v8f;
typedef __attribute__((ext_vector_type(4)))  int      v4i;

#define DIM_      1024
#define HEADS_    16
#define HD_       64      // head dim
#define B_        2
#define N_        2048
#define MROWS_    (B_ * N_)   // 4096 flattened rows

// ------------------------- WMMA fragment helpers ---------------------------
// A-matrix 16x32 f16: lanes 0-15 -> M=lane, K 0..15 contiguous
//                     lanes 16-31 -> M=lane-16, K 16..31 contiguous
__device__ __forceinline__ v16h frag_a(const _Float16* base, int ld, int k0, int lane) {
  int m  = lane & 15;
  int kc = (lane >> 4) << 4;              // 0 or 16
  const _Float16* p = base + m * ld + k0 + kc;
  v8h lo = *(const v8h*)(p);
  v8h hi = *(const v8h*)(p + 8);
  v16h r;
#pragma unroll
  for (int i = 0; i < 8; ++i) { r[i] = lo[i]; r[i + 8] = hi[i]; }
  return r;
}

// B-matrix 32x16 f16: lane%16 = N, lane/16 selects K half; 16 contiguous K
// values per lane when source is stored [N][K] row-major.
__device__ __forceinline__ v16h frag_b(const _Float16* base, int ld, int n0, int k0, int lane) {
  int n  = n0 + (lane & 15);
  int kc = (lane >> 4) << 4;              // 0 or 16
  return *(const v16h*)(base + n * ld + k0 + kc);
}

__device__ __forceinline__ v8f wmma_f16(v16h a, v16h b, v8f c) {
  return __builtin_amdgcn_wmma_f32_16x16x32_f16(false, a, false, b, (short)0, c,
                                                false, false);
}

// ----------------- async global->LDS copy (CDNA5 path) ---------------------
#if __has_builtin(__builtin_amdgcn_global_load_async_to_lds_b128) && \
    __has_builtin(__builtin_amdgcn_s_wait_asynccnt)
#define USE_ASYNC_LDS 1
#else
#define USE_ASYNC_LDS 0
#endif

__device__ __forceinline__ void cp_async_b128(const _Float16* g, _Float16* l) {
#if USE_ASYNC_LDS
  // builtin signature wants v4i pointers: global (AS1) source, LDS (AS3) dest.
  // Route through uintptr_t: strips const, and the generic LDS address
  // truncates to the AS3 offset per the flat-aperture rule.
  __builtin_amdgcn_global_load_async_to_lds_b128(
      (__attribute__((address_space(1))) v4i*)(uintptr_t)g,
      (__attribute__((address_space(3))) v4i*)(uint32_t)(uintptr_t)l, 0, 0);
#else
  *(v8h*)l = *(const v8h*)g;   // synchronous 16B fallback
#endif
}

__device__ __forceinline__ void wait_async_lds() {
#if USE_ASYNC_LDS
  __builtin_amdgcn_s_wait_asynccnt(0);
#endif
}

// ---------------------------- fp32 -> f16 convert --------------------------
__global__ void mha_cvt_f16(const float* __restrict__ src,
                            _Float16* __restrict__ dst, int n4) {
  int i = blockIdx.x * blockDim.x + threadIdx.x;
  if (i >= n4) return;
  float4 x = ((const float4*)src)[i];
  v4h o;
  o[0] = (_Float16)x.x; o[1] = (_Float16)x.y;
  o[2] = (_Float16)x.z; o[3] = (_Float16)x.w;
  ((v4h*)dst)[i] = o;
}

// --------------------- QKV projection: X @ W^T + b -------------------------
// X [4096,1024] f16 row-major, W [1024,1024] f16 row-major (NT gemm:
// out[m,n] = sum_k X[m,k] * W[n,k]).  Each wave owns a 32x64 C tile.
// Output is written head-major [B,H,N,64] f16; Q gets scale = 1/sqrt(64).
__global__ __launch_bounds__(256) void mha_qkv_proj(
    const _Float16* __restrict__ X, const _Float16* __restrict__ W,
    const float* __restrict__ bias, _Float16* __restrict__ dst, float scale) {
  int lane = threadIdx.x & 31;
  int wg   = blockIdx.x * 8 + (threadIdx.x >> 5);
  int tn   = wg & 15;                 // 16 column tiles of 64
  int tm   = wg >> 4;                 // 128 row tiles of 32
  int row0 = tm * 32, col0 = tn * 64;

  v8f c[2][4];
#pragma unroll
  for (int mi = 0; mi < 2; ++mi)
#pragma unroll
    for (int j = 0; j < 4; ++j)
#pragma unroll
      for (int r = 0; r < 8; ++r) c[mi][j][r] = 0.f;

  for (int k0 = 0; k0 < DIM_; k0 += 32) {
    v16h a0 = frag_a(X + (size_t)row0 * DIM_, DIM_, k0, lane);
    v16h a1 = frag_a(X + (size_t)(row0 + 16) * DIM_, DIM_, k0, lane);
#pragma unroll
    for (int j = 0; j < 4; ++j) {
      v16h b = frag_b(W, DIM_, col0 + j * 16, k0, lane);
      c[0][j] = wmma_f16(a0, b, c[0][j]);
      c[1][j] = wmma_f16(a1, b, c[1][j]);
    }
  }

  int nlane = lane & 15, mhalf = (lane >> 4) << 3;
#pragma unroll
  for (int j = 0; j < 4; ++j) {
    int n = col0 + j * 16 + nlane;
    float bn = bias[n];
    int h = n >> 6, d = n & 63;
#pragma unroll
    for (int mi = 0; mi < 2; ++mi)
#pragma unroll
      for (int r = 0; r < 8; ++r) {
        int m    = row0 + mi * 16 + mhalf + r;
        int bb   = m >> 11;            // N_ = 2048 rows per batch
        int nrow = m & (N_ - 1);
        dst[(((size_t)bb * HEADS_ + h) * N_ + nrow) * HD_ + d] =
            (_Float16)((c[mi][j][r] + bn) * scale);
      }
  }
}

// --------------------------- flash attention -------------------------------
// Grid (N/128, H, B); 8 waves per block, each wave owns 16 query rows so the
// online softmax stays wave-local.  K tile and V tile are double-buffered in
// LDS and streamed with GLOBAL_LOAD_ASYNC_TO_LDS_B128 (ASYNCcnt), overlapping
// the next block's HBM/L2 traffic with the current block's WMMAs.
#define KSTR 80   // padded f16 stride: 160B = 32B-aligned rows, spreads banks

// copy one 64x64 f16 tile (row-major, ld=64) into LDS with stride KSTR
__device__ __forceinline__ void stage_tile(const _Float16* gsrc,
                                           _Float16* lds, int tid) {
#pragma unroll
  for (int c = tid; c < 512; c += 256) {        // 512 x 16B chunks
    int row = c >> 3;
    int off = (c & 7) << 3;                     // halves
    cp_async_b128(gsrc + row * HD_ + off, lds + row * KSTR + off);
  }
}

__global__ __launch_bounds__(256) void mha_flash_attn(
    const _Float16* __restrict__ Qh, const _Float16* __restrict__ Kh,
    const _Float16* __restrict__ Vh, _Float16* __restrict__ ctx) {
  __shared__ __align__(32) _Float16 sK [2][64 * KSTR];     // [key][d]
  __shared__ __align__(32) _Float16 sVr[2][64 * KSTR];     // [key][d] staged
  __shared__ __align__(32) _Float16 sVt[2][64 * KSTR];     // [d][key] transposed
  __shared__ __align__(32) _Float16 sP [8 * 16 * KSTR];    // per-wave P tiles

  int tid  = threadIdx.x;
  int lane = tid & 31;
  int wid  = tid >> 5;
  int qb = blockIdx.x, h = blockIdx.y, b = blockIdx.z;
  size_t bh = (size_t)b * HEADS_ + h;
  int q0 = qb * 128 + wid * 16;

  const _Float16* Qbase = Qh + (bh * N_ + q0) * HD_;   // Q pre-scaled by 1/8
  v16h qa0 = frag_a(Qbase, HD_, 0, lane);
  v16h qa1 = frag_a(Qbase, HD_, 32, lane);

  v8f o[4];
  float mrow[8], lrow[8];
#pragma unroll
  for (int dt = 0; dt < 4; ++dt)
#pragma unroll
    for (int r = 0; r < 8; ++r) o[dt][r] = 0.f;
#pragma unroll
  for (int r = 0; r < 8; ++r) { mrow[r] = -1e30f; lrow[r] = 0.f; }

  _Float16* Pl = &sP[wid * 16 * KSTR];
  const _Float16* Kg = Kh + bh * N_ * HD_;
  const _Float16* Vg = Vh + bh * N_ * HD_;

  // prologue: issue async loads for the first tile pair
  stage_tile(Kg, sK[0], tid);
  stage_tile(Vg, sVr[0], tid);

  for (int it = 0; it < N_ / 64; ++it) {
    int buf = it & 1;

    wait_async_lds();          // this wave's outstanding async copies done
    __syncthreads();           // all waves' tile data visible in LDS

    if (it + 1 < N_ / 64) {    // stream next tile pair into the other buffer
      stage_tile(Kg + (size_t)(it + 1) * 64 * HD_, sK[buf ^ 1], tid);
      stage_tile(Vg + (size_t)(it + 1) * 64 * HD_, sVr[buf ^ 1], tid);
    }

    // LDS->LDS transpose of V tile: sVr[key][d] -> sVt[d][key]
    {
      int d  = tid & 63;
      int kg = (tid >> 6) << 4;
#pragma unroll
      for (int i = 0; i < 16; ++i)
        sVt[buf][d * KSTR + kg + i] = sVr[buf][(kg + i) * KSTR + d];
    }
    __syncthreads();           // transposed V visible

    // --- S = Q K^T (softmax scale already folded into Q) ---
    v8f s[4];
#pragma unroll
    for (int nt = 0; nt < 4; ++nt) {
      v8f acc;
#pragma unroll
      for (int r = 0; r < 8; ++r) acc[r] = 0.f;
      acc = wmma_f16(qa0, frag_b(sK[buf], KSTR, nt * 16, 0, lane), acc);
      acc = wmma_f16(qa1, frag_b(sK[buf], KSTR, nt * 16, 32, lane), acc);
      s[nt] = acc;
    }

    // --- wave-local online softmax (rows live in 16-lane groups) ---
#pragma unroll
    for (int r = 0; r < 8; ++r) {
      float mx = fmaxf(fmaxf(s[0][r], s[1][r]), fmaxf(s[2][r], s[3][r]));
#pragma unroll
      for (int off = 1; off < 16; off <<= 1)
        mx = fmaxf(mx, __shfl_xor(mx, off, 32));
      float mnew  = fmaxf(mrow[r], mx);
      float alpha = __expf(mrow[r] - mnew);
      mrow[r] = mnew;
      float rs = 0.f;
#pragma unroll
      for (int nt = 0; nt < 4; ++nt) {
        float p = __expf(s[nt][r] - mnew);
        s[nt][r] = p;
        rs += p;
      }
#pragma unroll
      for (int off = 1; off < 16; off <<= 1)
        rs += __shfl_xor(rs, off, 32);
      lrow[r] = lrow[r] * alpha + rs;
#pragma unroll
      for (int dt = 0; dt < 4; ++dt) o[dt][r] *= alpha;
    }

    // --- reshape P through per-wave LDS (C layout -> A fragment layout) ---
    {
      int nl = lane & 15, mh = (lane >> 4) << 3;
#pragma unroll
      for (int nt = 0; nt < 4; ++nt)
#pragma unroll
        for (int r = 0; r < 8; ++r)
          Pl[(mh + r) * KSTR + nt * 16 + nl] = (_Float16)s[nt][r];
    }
    v16h pa0 = frag_a(Pl, KSTR, 0, lane);
    v16h pa1 = frag_a(Pl, KSTR, 32, lane);

    // --- O += P V ---
#pragma unroll
    for (int dt = 0; dt < 4; ++dt) {
      o[dt] = wmma_f16(pa0, frag_b(sVt[buf], KSTR, dt * 16, 0, lane), o[dt]);
      o[dt] = wmma_f16(pa1, frag_b(sVt[buf], KSTR, dt * 16, 32, lane), o[dt]);
    }
  }

  // --- normalize and write context [B,N,DIM] f16 ---
  int nl = lane & 15, mh = (lane >> 4) << 3;
#pragma unroll
  for (int dt = 0; dt < 4; ++dt)
#pragma unroll
    for (int r = 0; r < 8; ++r) {
      int row = q0 + mh + r;
      int col = h * HD_ + dt * 16 + nl;
      ctx[((size_t)b * N_ + row) * DIM_ + col] = (_Float16)(o[dt][r] / lrow[r]);
    }
}

// ----------------------- output projection (fp32 out) ----------------------
__global__ __launch_bounds__(256) void mha_out_proj(
    const _Float16* __restrict__ X, const _Float16* __restrict__ W,
    const float* __restrict__ bias, float* __restrict__ out) {
  int lane = threadIdx.x & 31;
  int wg   = blockIdx.x * 8 + (threadIdx.x >> 5);
  int tn   = wg & 15;
  int tm   = wg >> 4;
  int row0 = tm * 32, col0 = tn * 64;

  v8f c[2][4];
#pragma unroll
  for (int mi = 0; mi < 2; ++mi)
#pragma unroll
    for (int j = 0; j < 4; ++j)
#pragma unroll
      for (int r = 0; r < 8; ++r) c[mi][j][r] = 0.f;

  for (int k0 = 0; k0 < DIM_; k0 += 32) {
    v16h a0 = frag_a(X + (size_t)row0 * DIM_, DIM_, k0, lane);
    v16h a1 = frag_a(X + (size_t)(row0 + 16) * DIM_, DIM_, k0, lane);
#pragma unroll
    for (int j = 0; j < 4; ++j) {
      v16h b = frag_b(W, DIM_, col0 + j * 16, k0, lane);
      c[0][j] = wmma_f16(a0, b, c[0][j]);
      c[1][j] = wmma_f16(a1, b, c[1][j]);
    }
  }

  int nlane = lane & 15, mhalf = (lane >> 4) << 3;
#pragma unroll
  for (int j = 0; j < 4; ++j) {
    int n = col0 + j * 16 + nlane;
    float bn = bias[n];
#pragma unroll
    for (int mi = 0; mi < 2; ++mi)
#pragma unroll
      for (int r = 0; r < 8; ++r) {
        int m = row0 + mi * 16 + mhalf + r;
        out[(size_t)m * DIM_ + n] = c[mi][j][r] + bn;
      }
  }
}

// ------------------------------- launcher ----------------------------------
extern "C" void kernel_launch(void* const* d_in, const int* in_sizes, int n_in,
                              void* d_out, int out_size, void* d_ws, size_t ws_size,
                              hipStream_t stream) {
  (void)in_sizes; (void)n_in; (void)out_size; (void)ws_size;

  const float* q_in = (const float*)d_in[0];
  const float* k_in = (const float*)d_in[1];
  const float* v_in = (const float*)d_in[2];
  const float* Wq   = (const float*)d_in[3];
  const float* bq   = (const float*)d_in[4];
  const float* Wk   = (const float*)d_in[5];
  const float* bk   = (const float*)d_in[6];
  const float* Wv   = (const float*)d_in[7];
  const float* bv   = (const float*)d_in[8];
  const float* Wo   = (const float*)d_in[9];
  const float* bo   = (const float*)d_in[10];

  char* ws = (char*)d_ws;
  const size_t MB = 1ull << 20;
  _Float16* xq  = (_Float16*)(ws + 0 * MB);    // 8 MB  [4096,1024] f16
  _Float16* xk  = (_Float16*)(ws + 8 * MB);
  _Float16* xv  = (_Float16*)(ws + 16 * MB);
  _Float16* wqh = (_Float16*)(ws + 24 * MB);   // 2 MB each
  _Float16* wkh = (_Float16*)(ws + 26 * MB);
  _Float16* wvh = (_Float16*)(ws + 28 * MB);
  _Float16* woh = (_Float16*)(ws + 30 * MB);
  _Float16* Qh  = (_Float16*)(ws + 32 * MB);   // 8 MB each [B,H,N,64]
  _Float16* Kh  = (_Float16*)(ws + 40 * MB);
  _Float16* Vh  = (_Float16*)(ws + 48 * MB);
  _Float16* ctx = (_Float16*)(ws + 56 * MB);   // 8 MB [4096,1024]

  const int NX = MROWS_ * DIM_;   // 4M elems
  const int NW = DIM_ * DIM_;     // 1M elems
  dim3 blk(256);

  mha_cvt_f16<<<(NX / 4 + 255) / 256, blk, 0, stream>>>(q_in, xq, NX / 4);
  mha_cvt_f16<<<(NX / 4 + 255) / 256, blk, 0, stream>>>(k_in, xk, NX / 4);
  mha_cvt_f16<<<(NX / 4 + 255) / 256, blk, 0, stream>>>(v_in, xv, NX / 4);
  mha_cvt_f16<<<(NW / 4 + 255) / 256, blk, 0, stream>>>(Wq, wqh, NW / 4);
  mha_cvt_f16<<<(NW / 4 + 255) / 256, blk, 0, stream>>>(Wk, wkh, NW / 4);
  mha_cvt_f16<<<(NW / 4 + 255) / 256, blk, 0, stream>>>(Wv, wvh, NW / 4);
  mha_cvt_f16<<<(NW / 4 + 255) / 256, blk, 0, stream>>>(Wo, woh, NW / 4);

  // 2048 waves of 32x64 tiles -> 256 blocks of 8 waves
  mha_qkv_proj<<<256, blk, 0, stream>>>(xq, wqh, bq, Qh, 0.125f); // 1/sqrt(64)
  mha_qkv_proj<<<256, blk, 0, stream>>>(xk, wkh, bk, Kh, 1.0f);
  mha_qkv_proj<<<256, blk, 0, stream>>>(xv, wvh, bv, Vh, 1.0f);

  mha_flash_attn<<<dim3(N_ / 128, HEADS_, B_), blk, 0, stream>>>(Qh, Kh, Vh, ctx);

  mha_out_proj<<<256, blk, 0, stream>>>(ctx, woh, bo, (float*)d_out);
}
